// TransformerLayer_45767171506282
// MI455X (gfx1250) — matrix operand order repaired and checked by reading, not compiled, not run on hardware
//
#include <hip/hip_runtime.h>
#include <hip/hip_bf16.h>
#include <math.h>

// ---------------------------------------------------------------------------
// Swin cross-attention transformer layer for MI455X (gfx1250, wave32, WMMA).
// All matmuls via v_wmma_f32_16x16x32_f16, LDS tiles 128x256x32, double-
// buffered with GLOBAL_LOAD_ASYNC_TO_LDS (ASYNCcnt) when available.
// ~155 GFLOP total vs ~600 MB scratch traffic @23.3 TB/s => compute bound;
// optimize for WMMA pipe utilization (16 wmma : 16 ds_load_b128 per wave/step).
// ---------------------------------------------------------------------------

typedef _Float16 f16;
typedef _Float16 v16h __attribute__((ext_vector_type(16)));
typedef _Float16 v8h  __attribute__((ext_vector_type(8)));
typedef float    v8f  __attribute__((ext_vector_type(8)));
typedef int      v4i32 __attribute__((vector_size(16)));

#define LN_EPS 1e-5f

// Problem dims (fixed by reference setup_inputs)
#define BATCH  8
#define HH     64
#define WWID   64
#define DIM    256
#define NWIN   32      // b * ns * ns
#define LW     1024    // (64/2)*(64/2)
#define MROWS  32768   // b*h*w
#define DHID   2048    // 2*d*4
#define DCAT   512     // 2*d

// GEMM tiling
#define BM  128
#define BN  256
#define BK  32
#define LDT 40         // LDS row stride (f16): 80B rows, conflict-padded, 16B aligned

#if defined(__HIP_DEVICE_COMPILE__) &&                                        \
    __has_builtin(__builtin_amdgcn_global_load_async_to_lds_b128) &&          \
    __has_builtin(__builtin_amdgcn_s_wait_asynccnt)
#define HAVE_ASYNC_LDS 1
#else
#define HAVE_ASYNC_LDS 0
#endif

__device__ __forceinline__ void async_cp16(const f16* g, f16* l) {
#if HAVE_ASYNC_LDS
  __builtin_amdgcn_global_load_async_to_lds_b128(
      (__attribute__((address_space(1))) v4i32*)(g),
      (__attribute__((address_space(3))) v4i32*)(l), 0, 0);
#else
  (void)g; (void)l;
#endif
}

// ---------------------------------------------------------------------------
__global__ __launch_bounds__(256)
void f32_to_f16_kernel(const float* __restrict__ src, f16* __restrict__ dst, int n) {
  int i = (blockIdx.x * blockDim.x + threadIdx.x) * 4;
  if (i + 3 < n) {
    float4 v = *(const float4*)(src + i);
    dst[i + 0] = (f16)v.x; dst[i + 1] = (f16)v.y;
    dst[i + 2] = (f16)v.z; dst[i + 3] = (f16)v.w;
  } else {
    for (; i < n; ++i) dst[i] = (f16)src[i];
  }
}

// ---------------------------------------------------------------------------
// Fragment loaders matching the CDNA5 wave32 WMMA VGPR layouts.
__device__ __forceinline__ v16h frag_cat(v8h lo, v8h hi) {
  return __builtin_shufflevector(lo, hi, 0,1,2,3,4,5,6,7,8,9,10,11,12,13,14,15);
}

// A 16x32 f16: lane<16 -> row=lane,    K {0..7} then {16..23}
//              lane>=16 -> row=lane-16, K {8..15} then {24..31}
__device__ __forceinline__ v16h load_frag_a(const f16* __restrict__ tile, int lane) {
  int half = lane >> 4;
  int r    = lane & 15;
  const f16* p = tile + r * LDT + half * 8;
  v8h lo = *(const v8h*)(p);
  v8h hi = *(const v8h*)(p + 16);
  return frag_cat(lo, hi);
}

// B 32x16 f16 stored in LDS as [n][k]: lane<16 -> col=lane,    K {0..7} then {16..23}
//                                      lane>=16 -> col=lane-16, K {8..15} then {24..31}
__device__ __forceinline__ v16h load_frag_b(const f16* __restrict__ tile, int lane) {
  int half = lane >> 4;
  int n    = lane & 15;
  const f16* p = tile + n * LDT + half * 8;
  v8h lo = *(const v8h*)(p);
  v8h hi = *(const v8h*)(p + 16);
  return frag_cat(lo, hi);
}

__device__ __forceinline__ float gelu_exact(float x) {
  return 0.5f * x * (1.0f + erff(x * 0.70710678118654752f));
}

// ---------------------------------------------------------------------------
// Generic NT GEMM: C[M,N] = A[M,K] * B[N,K]^T, batched via blockIdx.z.
// EPI: 0 = f32 store, 1 = f16 store, 2 = GELU->f16, 3 = scale+mask -> f32
// Requires M%128==0, N%256==0, K%32==0 (true for every GEMM in this layer).
template<int EPI>
__global__ __launch_bounds__(256)
void wmma_gemm_nt(const f16* __restrict__ A, const f16* __restrict__ B,
                  void* __restrict__ Cv, int M, int N, int K,
                  long long sA, long long sB, long long sC,
                  float scale, const float* __restrict__ mask, int maskMod) {
  __shared__ f16 sAm[2][BM * LDT];   // 2 x 10240 B
  __shared__ f16 sBm[2][BN * LDT];   // 2 x 20480 B   (total 60 KB)

  const int z = blockIdx.z;
  const f16* Ag = A + (long long)z * sA;
  const f16* Bg = B + (long long)z * sB;

  const int bm = blockIdx.y * BM;
  const int bn = blockIdx.x * BN;
  const int tid  = threadIdx.x;
  const int lane = tid & 31;
  const int wave = tid >> 5;
  const int wm = (wave >> 2) * 64;  // wave row offset: 0 / 64
  const int wn = (wave & 3) * 64;   // wave col offset: 0..192

  v8f acc[4][4];
#pragma unroll
  for (int mt = 0; mt < 4; ++mt)
#pragma unroll
    for (int nt = 0; nt < 4; ++nt) acc[mt][nt] = (v8f){0,0,0,0,0,0,0,0};

  // Cooperative tile-load mapping:
  //   A tile 128x32: thread t -> row t>>1, 16 f16 at col (t&1)*16  (2 x b128)
  //   B tile 256x32: thread t -> row t,    32 f16                  (4 x b128)
  const int ar = tid >> 1;
  const int ac = (tid & 1) * 16;
  const int nk = K / BK;

#if HAVE_ASYNC_LDS
  // ---- double-buffered pipeline with async global->LDS copies -------------
  {
    const f16* ga = Ag + (long long)(bm + ar) * K + ac;
    const f16* gb = Bg + (long long)(bn + tid) * K;
    f16* la = &sAm[0][ar * LDT + ac];
    f16* lb = &sBm[0][tid * LDT];
    async_cp16(ga, la);           async_cp16(ga + 8, la + 8);
    async_cp16(gb, lb);           async_cp16(gb + 8, lb + 8);
    async_cp16(gb + 16, lb + 16); async_cp16(gb + 24, lb + 24);
  }
  for (int kt = 0; kt < nk; ++kt) {
    const int cur = kt & 1;
    if (kt + 1 < nk) {
      const int k1 = (kt + 1) * BK;
      const f16* ga = Ag + (long long)(bm + ar) * K + k1 + ac;
      const f16* gb = Bg + (long long)(bn + tid) * K + k1;
      f16* la = &sAm[cur ^ 1][ar * LDT + ac];
      f16* lb = &sBm[cur ^ 1][tid * LDT];
      async_cp16(ga, la);           async_cp16(ga + 8, la + 8);
      async_cp16(gb, lb);           async_cp16(gb + 8, lb + 8);
      async_cp16(gb + 16, lb + 16); async_cp16(gb + 24, lb + 24);
      __builtin_amdgcn_s_wait_asynccnt(6);   // current tile's 6 loads are done
    } else {
      __builtin_amdgcn_s_wait_asynccnt(0);
    }
    __syncthreads();                          // all waves' tile data visible

    v16h af[4], bf[4];
#pragma unroll
    for (int mt = 0; mt < 4; ++mt)
      af[mt] = load_frag_a(&sAm[cur][(wm + mt * 16) * LDT], lane);
#pragma unroll
    for (int nt = 0; nt < 4; ++nt)
      bf[nt] = load_frag_b(&sBm[cur][(wn + nt * 16) * LDT], lane);
#pragma unroll
    for (int mt = 0; mt < 4; ++mt)
#pragma unroll
      for (int nt = 0; nt < 4; ++nt)
        acc[mt][nt] = __builtin_amdgcn_wmma_f32_16x16x32_f16(
            false, af[mt], false, bf[nt], (short)0, acc[mt][nt], false, false);
    __syncthreads();        // protect buffer `cur` before next iter's async writes
  }
#else
  // ---- synchronous fallback (register-staged, single buffer) --------------
  for (int kt = 0; kt < nk; ++kt) {
    const int k0 = kt * BK;
    const uint4* ga = (const uint4*)(Ag + (long long)(bm + ar) * K + k0 + ac);
    const uint4* gb = (const uint4*)(Bg + (long long)(bn + tid) * K + k0);
    uint4 a0 = ga[0], a1 = ga[1];
    uint4 b0 = gb[0], b1 = gb[1], b2 = gb[2], b3 = gb[3];
    *(uint4*)(&sAm[0][ar * LDT + ac + 0]) = a0;
    *(uint4*)(&sAm[0][ar * LDT + ac + 8]) = a1;
    *(uint4*)(&sBm[0][tid * LDT + 0])  = b0;
    *(uint4*)(&sBm[0][tid * LDT + 8])  = b1;
    *(uint4*)(&sBm[0][tid * LDT + 16]) = b2;
    *(uint4*)(&sBm[0][tid * LDT + 24]) = b3;
    __syncthreads();

    v16h af[4], bf[4];
#pragma unroll
    for (int mt = 0; mt < 4; ++mt)
      af[mt] = load_frag_a(&sAm[0][(wm + mt * 16) * LDT], lane);
#pragma unroll
    for (int nt = 0; nt < 4; ++nt)
      bf[nt] = load_frag_b(&sBm[0][(wn + nt * 16) * LDT], lane);
#pragma unroll
    for (int mt = 0; mt < 4; ++mt)
#pragma unroll
      for (int nt = 0; nt < 4; ++nt)
        acc[mt][nt] = __builtin_amdgcn_wmma_f32_16x16x32_f16(
            false, af[mt], false, bf[nt], (short)0, acc[mt][nt], false, false);
    __syncthreads();
  }
#endif

  // C layout: c[r] at lane -> M = r + (lane>=16 ? 8:0), N = lane&15
  const int ln = lane & 15;
  const int mo = (lane >> 4) * 8;
#pragma unroll
  for (int mt = 0; mt < 4; ++mt) {
#pragma unroll
    for (int nt = 0; nt < 4; ++nt) {
#pragma unroll
      for (int r = 0; r < 8; ++r) {
        int gm = bm + wm + mt * 16 + mo + r;
        int gn = bn + wn + nt * 16 + ln;
        long long ci = (long long)z * sC + (long long)gm * N + gn;
        float v = acc[mt][nt][r];
        if (EPI == 0) {
          ((float*)Cv)[ci] = v;
        } else if (EPI == 1) {
          ((f16*)Cv)[ci] = (f16)v;
        } else if (EPI == 2) {
          ((f16*)Cv)[ci] = (f16)gelu_exact(v);
        } else { // EPI == 3: scores: v*scale + mask[z % maskMod][gm][gn]
          long long mi = (long long)(z % maskMod) * (long long)M * N +
                         (long long)gm * N + gn;
          ((float*)Cv)[ci] = v * scale + mask[mi];
        }
      }
    }
  }
}

// ---------------------------------------------------------------------------
// Shifted-window gather:  Q,K -> [win][l][c] ; V -> transposed [win][c][l]
__global__ __launch_bounds__(256)
void window_gather(const f16* __restrict__ Q, const f16* __restrict__ Kt,
                   const f16* __restrict__ V, f16* __restrict__ Qw,
                   f16* __restrict__ Kw, f16* __restrict__ Vwt) {
  int idx = blockIdx.x * blockDim.x + threadIdx.x;      // NWIN*LW*(DIM/8)
  if (idx >= NWIN * LW * (DIM / 8)) return;
  int c8  = idx & 31;
  int l   = (idx >> 5) & (LW - 1);
  int win = idx >> 15;
  int b = win >> 2, wy = (win >> 1) & 1, wx = win & 1;
  int i = l >> 5, j = l & 31;
  int y = (wy * 32 + i + 16) & 63;   // roll by -h/(2ns) = -16 (mod 64)
  int x = (wx * 32 + j + 16) & 63;
  long long src = ((long long)(b * 4096 + y * 64 + x)) * DIM + c8 * 8;
  long long dst = ((long long)win * LW + l) * DIM + c8 * 8;
  *(uint4*)(Qw + dst) = *(const uint4*)(Q + src);
  *(uint4*)(Kw + dst) = *(const uint4*)(Kt + src);
  const f16* vs = V + src;
  long long vb = (long long)win * LW * DIM;
#pragma unroll
  for (int t = 0; t < 8; ++t) Vwt[vb + (long long)(c8 * 8 + t) * LW + l] = vs[t];
}

// Inverse: merge windows + roll back
__global__ __launch_bounds__(256)
void window_merge(const f16* __restrict__ Ow, f16* __restrict__ Mh) {
  int idx = blockIdx.x * blockDim.x + threadIdx.x;
  if (idx >= NWIN * LW * (DIM / 8)) return;
  int c8  = idx & 31;
  int l   = (idx >> 5) & (LW - 1);
  int win = idx >> 15;
  int b = win >> 2, wy = (win >> 1) & 1, wx = win & 1;
  int i = l >> 5, j = l & 31;
  int y = (wy * 32 + i + 16) & 63;
  int x = (wx * 32 + j + 16) & 63;
  long long dstp = ((long long)(b * 4096 + y * 64 + x)) * DIM + c8 * 8;
  long long srcp = ((long long)win * LW + l) * DIM + c8 * 8;
  *(uint4*)(Mh + dstp) = *(const uint4*)(Ow + srcp);
}

// ---------------------------------------------------------------------------
__global__ __launch_bounds__(256)
void softmax_rows(const float* __restrict__ S, f16* __restrict__ P, int ncols) {
  __shared__ float red[256];
  long long row = blockIdx.x;
  const float* x = S + row * ncols;
  int t = threadIdx.x;
  float m = -INFINITY;
  for (int c = t; c < ncols; c += 256) m = fmaxf(m, x[c]);
  red[t] = m; __syncthreads();
  for (int s = 128; s > 0; s >>= 1) {
    if (t < s) red[t] = fmaxf(red[t], red[t + s]);
    __syncthreads();
  }
  m = red[0]; __syncthreads();
  float sum = 0.f;
  for (int c = t; c < ncols; c += 256) sum += expf(x[c] - m);
  red[t] = sum; __syncthreads();
  for (int s = 128; s > 0; s >>= 1) {
    if (t < s) red[t] += red[t + s];
    __syncthreads();
  }
  float inv = 1.0f / red[0];
  for (int c = t; c < ncols; c += 256)
    P[row * ncols + c] = (f16)(expf(x[c] - m) * inv);
}

// ---------------------------------------------------------------------------
// LN over c=256; writes ln(msgW) into right half of concat buffer, copies
// f16(source) into left half.
__global__ __launch_bounds__(256)
void ln1_concat(const float* __restrict__ msg, const float* __restrict__ g,
                const float* __restrict__ bta, const float* __restrict__ src,
                f16* __restrict__ CH) {
  __shared__ float red[256];
  long long row = blockIdx.x;
  int t = threadIdx.x;
  float v = msg[row * DIM + t];
  red[t] = v; __syncthreads();
  for (int s = 128; s > 0; s >>= 1) { if (t < s) red[t] += red[t + s]; __syncthreads(); }
  float mu = red[0] * (1.0f / DIM); __syncthreads();
  float d = v - mu;
  red[t] = d * d; __syncthreads();
  for (int s = 128; s > 0; s >>= 1) { if (t < s) red[t] += red[t + s]; __syncthreads(); }
  float var = red[0] * (1.0f / DIM);
  float nv = d * rsqrtf(var + LN_EPS) * g[t] + bta[t];
  CH[row * DCAT + DIM + t] = (f16)nv;
  CH[row * DCAT + t]       = (f16)src[row * DIM + t];
}

__global__ __launch_bounds__(256)
void ln2_residual(const float* __restrict__ y, const float* __restrict__ g,
                  const float* __restrict__ bta, const float* __restrict__ src,
                  float* __restrict__ out) {
  __shared__ float red[256];
  long long row = blockIdx.x;
  int t = threadIdx.x;
  float v = y[row * DIM + t];
  red[t] = v; __syncthreads();
  for (int s = 128; s > 0; s >>= 1) { if (t < s) red[t] += red[t + s]; __syncthreads(); }
  float mu = red[0] * (1.0f / DIM); __syncthreads();
  float d = v - mu;
  red[t] = d * d; __syncthreads();
  for (int s = 128; s > 0; s >>= 1) { if (t < s) red[t] += red[t + s]; __syncthreads(); }
  float var = red[0] * (1.0f / DIM);
  float nv = d * rsqrtf(var + LN_EPS) * g[t] + bta[t];
  out[row * DIM + t] = src[row * DIM + t] + nv;
}

// ---------------------------------------------------------------------------
extern "C" void kernel_launch(void* const* d_in, const int* in_sizes, int n_in,
                              void* d_out, int out_size, void* d_ws, size_t ws_size,
                              hipStream_t stream) {
  const float* source = (const float*)d_in[0];
  const float* target = (const float*)d_in[1];
  const float* mask   = (const float*)d_in[2];
  const float* Wq     = (const float*)d_in[3];
  const float* Wk     = (const float*)d_in[4];
  const float* Wv     = (const float*)d_in[5];
  const float* Wm     = (const float*)d_in[6];
  const float* ln1_g  = (const float*)d_in[7];
  const float* ln1_b  = (const float*)d_in[8];
  const float* W1     = (const float*)d_in[9];
  const float* W2     = (const float*)d_in[10];
  const float* ln2_g  = (const float*)d_in[11];
  const float* ln2_b  = (const float*)d_in[12];

  // bump allocator over workspace (~607 MB total; deterministic layout)
  char* w = (char*)d_ws;
  auto alloc = [&](size_t bytes) -> void* {
    void* p = (void*)w;
    w += (bytes + 255) & ~(size_t)255;
    return p;
  };
  f16*   Sh   = (f16*)  alloc((size_t)MROWS * DIM * 2);
  f16*   Th   = (f16*)  alloc((size_t)MROWS * DIM * 2);
  f16*   Wqh  = (f16*)  alloc((size_t)DIM * DIM * 2);
  f16*   Wkh  = (f16*)  alloc((size_t)DIM * DIM * 2);
  f16*   Wvh  = (f16*)  alloc((size_t)DIM * DIM * 2);
  f16*   Wmh  = (f16*)  alloc((size_t)DIM * DIM * 2);
  f16*   W1h  = (f16*)  alloc((size_t)DHID * DCAT * 2);
  f16*   W2h  = (f16*)  alloc((size_t)DIM * DHID * 2);
  f16*   Qh   = (f16*)  alloc((size_t)MROWS * DIM * 2);
  f16*   Kh   = (f16*)  alloc((size_t)MROWS * DIM * 2);
  f16*   Vh   = (f16*)  alloc((size_t)MROWS * DIM * 2);
  f16*   Qw   = (f16*)  alloc((size_t)NWIN * LW * DIM * 2);
  f16*   Kw   = (f16*)  alloc((size_t)NWIN * LW * DIM * 2);
  f16*   Vwt  = (f16*)  alloc((size_t)NWIN * LW * DIM * 2);
  float* Sc   = (float*)alloc((size_t)NWIN * LW * LW * 4);
  f16*   P    = (f16*)  alloc((size_t)NWIN * LW * LW * 2);
  f16*   Ow   = (f16*)  alloc((size_t)NWIN * LW * DIM * 2);
  f16*   Mh   = (f16*)  alloc((size_t)MROWS * DIM * 2);
  float* MsgW = (float*)alloc((size_t)MROWS * DIM * 4);
  f16*   CH   = (f16*)  alloc((size_t)MROWS * DCAT * 2);
  f16*   Hid  = (f16*)  alloc((size_t)MROWS * DHID * 2);
  float* Out2 = (float*)alloc((size_t)MROWS * DIM * 4);

  auto conv = [&](const float* s, f16* d, int n) {
    f32_to_f16_kernel<<<(n / 4 + 255) / 256, 256, 0, stream>>>(s, d, n);
  };
  conv(source, Sh, MROWS * DIM);
  conv(target, Th, MROWS * DIM);
  conv(Wq, Wqh, DIM * DIM);
  conv(Wk, Wkh, DIM * DIM);
  conv(Wv, Wvh, DIM * DIM);
  conv(Wm, Wmh, DIM * DIM);
  conv(W1, W1h, DHID * DCAT);
  conv(W2, W2h, DIM * DHID);

  // Projections: [32768,256] = [32768,256] x [256,256]^T
  dim3 blk(256);
  dim3 gP(DIM / BN, MROWS / BM, 1);
  wmma_gemm_nt<1><<<gP, blk, 0, stream>>>(Sh, Wqh, Qh, MROWS, DIM, DIM, 0, 0, 0, 1.f, nullptr, 1);
  wmma_gemm_nt<1><<<gP, blk, 0, stream>>>(Th, Wkh, Kh, MROWS, DIM, DIM, 0, 0, 0, 1.f, nullptr, 1);
  wmma_gemm_nt<1><<<gP, blk, 0, stream>>>(Th, Wvh, Vh, MROWS, DIM, DIM, 0, 0, 0, 1.f, nullptr, 1);

  // Shifted-window gather (+ V transpose)
  {
    int n = NWIN * LW * (DIM / 8);
    window_gather<<<(n + 255) / 256, 256, 0, stream>>>(Qh, Kh, Vh, Qw, Kw, Vwt);
  }

  // Scores: per window [1024,1024] = Qw x Kw^T, *1/sqrt(256), + mask[win%4]
  {
    dim3 g(LW / BN, LW / BM, NWIN);
    wmma_gemm_nt<3><<<g, blk, 0, stream>>>(
        Qw, Kw, Sc, LW, LW, DIM,
        (long long)LW * DIM, (long long)LW * DIM, (long long)LW * LW,
        0.0625f, mask, 4);
  }

  // Softmax rows -> f16 probabilities
  softmax_rows<<<NWIN * LW, 256, 0, stream>>>(Sc, P, LW);

  // Ow: per window [1024,256] = P x Vwt^T  (Vwt stored [256,1024])
  {
    dim3 g(DIM / BN, LW / BM, NWIN);
    wmma_gemm_nt<1><<<g, blk, 0, stream>>>(
        P, Vwt, Ow, LW, DIM, LW,
        (long long)LW * LW, (long long)LW * DIM, (long long)LW * DIM,
        1.f, nullptr, 1);
  }

  // Merge windows + inverse shift
  {
    int n = NWIN * LW * (DIM / 8);
    window_merge<<<(n + 255) / 256, 256, 0, stream>>>(Ow, Mh);
  }

  // msgW = msg x Wm^T (f32 accum out)
  wmma_gemm_nt<0><<<gP, blk, 0, stream>>>(Mh, Wmh, MsgW, MROWS, DIM, DIM, 0, 0, 0, 1.f, nullptr, 1);

  // LN1 + concat [f16(source) | ln(msgW)]
  ln1_concat<<<MROWS, 256, 0, stream>>>(MsgW, ln1_g, ln1_b, source, CH);

  // hid = GELU(concat x W1^T): [32768,2048], K=512
  {
    dim3 g(DHID / BN, MROWS / BM, 1);
    wmma_gemm_nt<2><<<g, blk, 0, stream>>>(CH, W1h, Hid, MROWS, DHID, DCAT, 0, 0, 0, 1.f, nullptr, 1);
  }

  // out2 = hid x W2^T: [32768,256], K=2048
  {
    dim3 g(DIM / BN, MROWS / BM, 1);
    wmma_gemm_nt<0><<<g, blk, 0, stream>>>(Hid, W2h, Out2, MROWS, DIM, DHID, 0, 0, 0, 1.f, nullptr, 1);
  }

  // Final LN2 + residual
  ln2_residual<<<MROWS, 256, 0, stream>>>(Out2, ln2_g, ln2_b, source, (float*)d_out);
}
